// PointnetSAModuleMSG_55327768708592
// MI455X (gfx1250) — compile-verified, matrix operand order checked by reading, not run on hardware
//
#include <hip/hip_runtime.h>
#include <hip/hip_bf16.h>

// ---------------------------------------------------------------------------
// PointNet++ SA-MSG module for MI455X (gfx1250, wave32, WMMA)
// ---------------------------------------------------------------------------
#define BATCH   4
#define NPTS    8192
#define CIN     64
#define NPOINT  2048
#define OUT_FEAT_BASE (BATCH * NPOINT * 3)   // new_xyz floats before features

typedef _Float16 v16h __attribute__((ext_vector_type(16)));
typedef _Float16 v8h  __attribute__((ext_vector_type(8)));
typedef float    v8f  __attribute__((ext_vector_type(8)));

// Load a 16-half WMMA operand fragment from two contiguous 16-byte chunks:
// halves [0..7] at p, halves [8..15] at p+16 (K layout of 16x16x32 f16 WMMA).
__device__ __forceinline__ v16h ld_frag(const _Float16* p) {
  v8h lo = *(const v8h*)(p);
  v8h hi = *(const v8h*)(p + 16);
  return __builtin_shufflevector(lo, hi, 0, 1, 2, 3, 4, 5, 6, 7,
                                 8, 9, 10, 11, 12, 13, 14, 15);
}

// ===========================================================================
// Kernel 1: Furthest point sampling. One block per batch, 1024 threads,
// 8 points per thread held entirely in registers. The argmax winner's
// coordinates ride along the shuffle reduction so the serial loop has no
// dependent global loads.
// ===========================================================================
#define FPS_THREADS 1024
#define PPT (NPTS / FPS_THREADS)   // 8

__global__ __launch_bounds__(FPS_THREADS)
void fps_kernel(const float* __restrict__ xyz, int* __restrict__ fidx,
                float* __restrict__ newxyz) {
  const int b    = blockIdx.x;
  const int tid  = threadIdx.x;
  const int lane = tid & 31;
  const int wid  = tid >> 5;

  __shared__ float swv[32], swx[32], swy[32], swz[32];
  __shared__ int   swi[32];
  __shared__ float sbx, sby, sbz;

  float px[PPT], py[PPT], pz[PPT], mind[PPT];
  const float* base = xyz + (size_t)b * NPTS * 3;
#pragma unroll
  for (int i = 0; i < PPT; ++i) {
    int p = tid + i * FPS_THREADS;
    px[i] = base[p * 3 + 0];
    py[i] = base[p * 3 + 1];
    pz[i] = base[p * 3 + 2];
    mind[i] = 1e10f;
  }
  if (tid == 0) {
    sbx = base[0]; sby = base[1]; sbz = base[2];
    fidx[(size_t)b * NPOINT] = 0;
    float* nz = newxyz + (size_t)b * NPOINT * 3;
    nz[0] = base[0]; nz[1] = base[1]; nz[2] = base[2];
  }
  __syncthreads();

  for (int s = 1; s < NPOINT; ++s) {
    const float cx = sbx, cy = sby, cz = sbz;
    float bv = -1.0f, bx = 0.f, by = 0.f, bz = 0.f;
    int   bi = 0x7fffffff;
#pragma unroll
    for (int i = 0; i < PPT; ++i) {
      float dx = px[i] - cx, dy = py[i] - cy, dz = pz[i] - cz;
      float d  = dx * dx + dy * dy + dz * dz;
      float md = fminf(mind[i], d);
      mind[i]  = md;
      int p = tid + i * FPS_THREADS;
      if (md > bv || (md == bv && p < bi)) {
        bv = md; bi = p; bx = px[i]; by = py[i]; bz = pz[i];
      }
    }
    // wave reduction (carry coords)
#pragma unroll
    for (int off = 16; off > 0; off >>= 1) {
      float ov = __shfl_xor(bv, off);
      int   oi = __shfl_xor(bi, off);
      float ox = __shfl_xor(bx, off);
      float oy = __shfl_xor(by, off);
      float oz = __shfl_xor(bz, off);
      if (ov > bv || (ov == bv && oi < bi)) { bv = ov; bi = oi; bx = ox; by = oy; bz = oz; }
    }
    if (lane == 0) { swv[wid] = bv; swi[wid] = bi; swx[wid] = bx; swy[wid] = by; swz[wid] = bz; }
    __syncthreads();
    if (wid == 0) {
      bv = swv[lane]; bi = swi[lane]; bx = swx[lane]; by = swy[lane]; bz = swz[lane];
#pragma unroll
      for (int off = 16; off > 0; off >>= 1) {
        float ov = __shfl_xor(bv, off);
        int   oi = __shfl_xor(bi, off);
        float ox = __shfl_xor(bx, off);
        float oy = __shfl_xor(by, off);
        float oz = __shfl_xor(bz, off);
        if (ov > bv || (ov == bv && oi < bi)) { bv = ov; bi = oi; bx = ox; by = oy; bz = oz; }
      }
      if (lane == 0) {
        sbx = bx; sby = by; sbz = bz;
        fidx[(size_t)b * NPOINT + s] = bi;
        float* nz = newxyz + ((size_t)b * NPOINT + s) * 3;
        nz[0] = bx; nz[1] = by; nz[2] = bz;
      }
    }
    __syncthreads();
  }
}

// ===========================================================================
// Kernel 2: Ball query. One wave per query; ballot + prefix-popcount keeps
// hits in ascending index order (matches reference "first K by index").
// ===========================================================================
__global__ __launch_bounds__(256)
void ballq_kernel(const float* __restrict__ xyz, const float* __restrict__ newxyz,
                  int* __restrict__ gidx, float r2, int K) {
  const int lane = threadIdx.x & 31;
  const int wid  = threadIdx.x >> 5;
  const int q = blockIdx.x * 8 + wid;      // 8 waves per block
  const int b = q / NPOINT;
  const float* c = newxyz + (size_t)q * 3;
  const float cx = c[0], cy = c[1], cz = c[2];
  const float* pts = xyz + (size_t)b * NPTS * 3;
  int* out = gidx + (size_t)q * K;

  int cnt = 0, firstIdx = -1;
  for (int basep = 0; basep < NPTS && cnt < K; basep += 32) {
    int p = basep + lane;
    float dx = pts[p * 3 + 0] - cx;
    float dy = pts[p * 3 + 1] - cy;
    float dz = pts[p * 3 + 2] - cz;
    bool hit = (dx * dx + dy * dy + dz * dz) < r2;
    unsigned mask = (unsigned)__ballot(hit);
    int pos  = __popc(mask & ((1u << lane) - 1u));
    int slot = cnt + pos;
    if (hit && slot < K) out[slot] = p;
    if (firstIdx < 0 && mask) firstIdx = basep + (__ffs((int)mask) - 1);
    cnt += __popc(mask);
  }
  int pad = (firstIdx < 0) ? (NPTS - 1) : firstIdx;
  int filled = cnt < K ? cnt : K;
  for (int s = filled + lane; s < K; s += 32) out[s] = pad;
}

// ===========================================================================
// Kernel 3: gather + 3-layer MLP (WMMA f16->f32) + fused BN/ReLU + maxpool.
// 128 threads (4 waves), 64 columns per block. Columns = (m,k) pairs.
// Activations live in LDS column-major [col][Kpad] so that B-fragment loads
// and D-fragment stores are contiguous 16-byte LDS accesses.
// ===========================================================================
#define COLS 64
#define LDB  104   // halves per column row: 96 + 8 pad (208 B, 16-B aligned)

template<int CO, int KP>
__device__ __forceinline__
void layer_fc(const _Float16* __restrict__ Wb, const _Float16* __restrict__ Xs,
              _Float16* __restrict__ Xd, const float* __restrict__ sc,
              const float* __restrict__ sh, int lane, int ws) {
  const int col   = ws * 16 + (lane & 15);
  const int khalf = (lane < 16) ? 0 : 8;
  constexpr int NK = KP / 32;
  v16h bfrag[NK];
  const _Float16* xc = Xs + (size_t)col * LDB + khalf;
#pragma unroll
  for (int kk = 0; kk < NK; ++kk) bfrag[kk] = ld_frag(xc + kk * 32);

  const int rowA = (lane & 15);
#pragma unroll
  for (int t = 0; t < CO / 16; ++t) {
    v8f acc = {0.f, 0.f, 0.f, 0.f, 0.f, 0.f, 0.f, 0.f};
    const _Float16* wr = Wb + (size_t)(t * 16 + rowA) * KP + khalf;
#pragma unroll
    for (int kk = 0; kk < NK; ++kk) {
      acc = __builtin_amdgcn_wmma_f32_16x16x32_f16(false, ld_frag(wr + kk * 32),
                                                   false, bfrag[kk],
                                                   (short)0, acc, false, false);
    }
    const int rbase = t * 16 + ((lane < 16) ? 0 : 8);
    v8h o;
#pragma unroll
    for (int r = 0; r < 8; ++r) {
      const int row = rbase + r;
      o[r] = (_Float16)fmaxf(acc[r] * sc[row] + sh[row], 0.f);
    }
    *(v8h*)(Xd + (size_t)col * LDB + rbase) = o;   // contiguous ds_store_b128
  }
}

template<int CO, int KP>
__device__ __forceinline__
void layer_fc_max(const _Float16* __restrict__ Wb, const _Float16* __restrict__ Xs,
                  const float* __restrict__ sc, const float* __restrict__ sh,
                  float (*pmax)[128], int lane, int ws) {
  const int col   = ws * 16 + (lane & 15);
  const int khalf = (lane < 16) ? 0 : 8;
  constexpr int NK = KP / 32;
  v16h bfrag[NK];
  const _Float16* xc = Xs + (size_t)col * LDB + khalf;
#pragma unroll
  for (int kk = 0; kk < NK; ++kk) bfrag[kk] = ld_frag(xc + kk * 32);

  const int rowA = (lane & 15);
#pragma unroll
  for (int t = 0; t < CO / 16; ++t) {
    v8f acc = {0.f, 0.f, 0.f, 0.f, 0.f, 0.f, 0.f, 0.f};
    const _Float16* wr = Wb + (size_t)(t * 16 + rowA) * KP + khalf;
#pragma unroll
    for (int kk = 0; kk < NK; ++kk) {
      acc = __builtin_amdgcn_wmma_f32_16x16x32_f16(false, ld_frag(wr + kk * 32),
                                                   false, bfrag[kk],
                                                   (short)0, acc, false, false);
    }
    const int rbase = t * 16 + ((lane < 16) ? 0 : 8);
#pragma unroll
    for (int r = 0; r < 8; ++r) {
      const int row = rbase + r;
      float v = fmaxf(acc[r] * sc[row] + sh[row], 0.f);
      // max over the 16 columns held by this 16-lane half (in registers)
#pragma unroll
      for (int off = 1; off < 16; off <<= 1) v = fmaxf(v, __shfl_xor(v, off));
      if ((lane & 15) == 0) pmax[ws][row] = v;
    }
  }
}

__device__ __forceinline__
void load_layer(const float* __restrict__ W, const float* __restrict__ G,
                const float* __restrict__ Bt, const float* __restrict__ Mn,
                const float* __restrict__ Vr, int CO, int CI, int KP,
                _Float16* __restrict__ Wbuf, float* __restrict__ sc,
                float* __restrict__ sh, int tid) {
  for (int i = tid; i < CO * KP; i += 128) {
    int r = i / KP, k = i % KP;
    float w = (k < CI) ? W[(size_t)r * CI + k] : 0.f;
    Wbuf[i] = (_Float16)w;
  }
  for (int i = tid; i < CO; i += 128) {
    float s = G[i] * rsqrtf(Vr[i] + 1e-5f);
    sc[i] = s;
    sh[i] = Bt[i] - Mn[i] * s;
  }
}

template<int SCALE>
__global__ __launch_bounds__(128)
void mlp_kernel(const float* __restrict__ xyz, const float* __restrict__ feats,
                const float* __restrict__ newxyz, const int* __restrict__ gidx,
                const float* W1, const float* G1, const float* B1, const float* M1, const float* V1,
                const float* W2, const float* G2, const float* B2, const float* M2, const float* V2,
                const float* W3, const float* G3, const float* B3, const float* M3, const float* V3,
                float* __restrict__ out) {
  constexpr int KS  = SCALE ? 32 : 16;        // samples per center
  constexpr int MPB = COLS / KS;              // centers per block (4 or 2)
  constexpr int CO1 = 64;
  constexpr int CO2 = SCALE ? 96 : 64;
  constexpr int CO3 = 128;
  constexpr int KP1 = 96;                     // 67 padded to 96
  constexpr int KP2 = 64;
  constexpr int KP3 = SCALE ? 96 : 64;

  __shared__ _Float16 Wbuf[128 * 96];         // max layer weight tile (f16)
  __shared__ _Float16 Xa[COLS * LDB];         // activations, column-major
  __shared__ _Float16 Xb[COLS * LDB];
  __shared__ float sc[128], sh[128];
  __shared__ int   colgi[COLS];
  __shared__ float colcx[COLS], colcy[COLS], colcz[COLS];
  __shared__ float pmax[4][128];

  const int tid  = threadIdx.x;
  const int lane = tid & 31;
  const int ws   = tid >> 5;
  constexpr int MB = NPOINT / MPB;            // blocks per batch
  const int b  = blockIdx.x / MB;
  const int m0 = (blockIdx.x % MB) * MPB;

  // ---- per-column gather metadata ----
  if (tid < COLS) {
    const int c = tid;
    const int m = m0 + c / KS;
    const int k = c % KS;
    colgi[c] = gidx[((size_t)(b * NPOINT + m)) * KS + k];
    const float* nc = newxyz + (size_t)(b * NPOINT + m) * 3;
    colcx[c] = nc[0]; colcy[c] = nc[1]; colcz[c] = nc[2];
  }
  __syncthreads();

  // ---- gather input tile (column-major): ch 0..2 recentered xyz,
  //      3..66 features, 67..95 zero-padded K ----
  for (int i = tid; i < COLS * 96; i += 128) {
    const int c = i / 96, row = i % 96;
    float v = 0.f;
    const int gi = colgi[c];
    if (row < 3) {
      const float* p = xyz + ((size_t)b * NPTS + gi) * 3;
      v = p[row] - (row == 0 ? colcx[c] : (row == 1 ? colcy[c] : colcz[c]));
    } else if (row < 67) {
      v = feats[((size_t)b * CIN + (row - 3)) * NPTS + gi];
    }
    Xa[(size_t)c * LDB + row] = (_Float16)v;
  }

  // ---- layer 1 ----
  load_layer(W1, G1, B1, M1, V1, CO1, 67, KP1, Wbuf, sc, sh, tid);
  __syncthreads();
  layer_fc<CO1, KP1>(Wbuf, Xa, Xb, sc, sh, lane, ws);
  __syncthreads();

  // ---- layer 2 ----
  load_layer(W2, G2, B2, M2, V2, CO2, 64, KP2, Wbuf, sc, sh, tid);
  __syncthreads();
  layer_fc<CO2, KP2>(Wbuf, Xb, Xa, sc, sh, lane, ws);
  __syncthreads();

  // ---- layer 3 + in-register maxpool ----
  load_layer(W3, G3, B3, M3, V3, CO3, CO2, KP3, Wbuf, sc, sh, tid);
  __syncthreads();
  layer_fc_max<CO3, KP3>(Wbuf, Xa, sc, sh, pmax, lane, ws);
  __syncthreads();

  // ---- combine per-wave partials and write [B,256,M] features ----
  constexpr int GS = 4 / MPB;                 // waves per center
  for (int i = tid; i < MPB * 128; i += 128) {
    const int ml = i >> 7;
    const int ch = i & 127;
    float v = pmax[ml * GS][ch];
#pragma unroll
    for (int g = 1; g < GS; ++g) v = fmaxf(v, pmax[ml * GS + g][ch]);
    out[OUT_FEAT_BASE +
        (((size_t)b * 256 + SCALE * 128 + ch) * NPOINT + (m0 + ml))] = v;
  }
}

// ===========================================================================
// Host launcher
// ===========================================================================
extern "C" void kernel_launch(void* const* d_in, const int* in_sizes, int n_in,
                              void* d_out, int out_size, void* d_ws, size_t ws_size,
                              hipStream_t stream) {
  const float* xyz   = (const float*)d_in[0];
  const float* feats = (const float*)d_in[1];
  // params pytree order: scale-major, layer-major, dict keys alphabetical:
  // beta, gamma, mean, var, w
  auto P = [&](int s, int l, int which) -> const float* {
    return (const float*)d_in[2 + (s * 3 + l) * 5 + which];
  };
  // which: 0=beta 1=gamma 2=mean 3=var 4=w

  int* fidx  = (int*)d_ws;
  int* gidx0 = fidx + BATCH * NPOINT;
  int* gidx1 = gidx0 + BATCH * NPOINT * 16;
  float* outf = (float*)d_out;

  fps_kernel<<<BATCH, FPS_THREADS, 0, stream>>>(xyz, fidx, outf);

  ballq_kernel<<<(BATCH * NPOINT) / 8, 256, 0, stream>>>(
      xyz, (const float*)d_out, gidx0, 0.1f * 0.1f, 16);
  ballq_kernel<<<(BATCH * NPOINT) / 8, 256, 0, stream>>>(
      xyz, (const float*)d_out, gidx1, 0.2f * 0.2f, 32);

  mlp_kernel<0><<<(BATCH * NPOINT) / 4, 128, 0, stream>>>(
      xyz, feats, (const float*)d_out, gidx0,
      P(0,0,4), P(0,0,1), P(0,0,0), P(0,0,2), P(0,0,3),
      P(0,1,4), P(0,1,1), P(0,1,0), P(0,1,2), P(0,1,3),
      P(0,2,4), P(0,2,1), P(0,2,0), P(0,2,2), P(0,2,3),
      outf);
  mlp_kernel<1><<<(BATCH * NPOINT) / 2, 128, 0, stream>>>(
      xyz, feats, (const float*)d_out, gidx1,
      P(1,0,4), P(1,0,1), P(1,0,0), P(1,0,2), P(1,0,3),
      P(1,1,4), P(1,1,1), P(1,1,0), P(1,1,2), P(1,1,3),
      P(1,2,4), P(1,2,1), P(1,2,0), P(1,2,2), P(1,2,3),
      outf);
}
